// GAT_28913719837049
// MI455X (gfx1250) — compile-verified, matrix operand order checked by reading, not run on hardware
//
#include <hip/hip_runtime.h>
#include <hip/hip_bf16.h>
#include <math.h>

#define NH 4
#define CC 48
#define D 192         // NH*CC
#define FEAT 128
#define GNUM 64
#define NEG_SLOPE 0.2f

typedef __attribute__((ext_vector_type(16))) __bf16 v16bf;
typedef __attribute__((ext_vector_type(8)))  __bf16 v8bf;
typedef __attribute__((ext_vector_type(8)))  float  v8f;

union V16U { v16bf v; struct { v8bf lo, hi; } p; };

// ---------- helpers ----------
__device__ __forceinline__ unsigned f2mono(float f) {
  unsigned u = __float_as_uint(f);
  return (u & 0x80000000u) ? ~u : (u | 0x80000000u);
}
__device__ __forceinline__ float mono2f(unsigned k) {
  return (k & 0x80000000u) ? __uint_as_float(k & 0x7fffffffu)
                           : __uint_as_float(~k);
}
__device__ __forceinline__ float elu_f(float v) { return v > 0.f ? v : (__expf(v) - 1.f); }

#define MONO_NEG_INF 0x007FFFFFu   // f2mono(-inf)

// ---------- f32 -> bf16 row-major convert ----------
__global__ void f32_to_bf16_kernel(const float* __restrict__ in, __bf16* __restrict__ outp,
                                   int n) {
  int i = blockIdx.x * blockDim.x + threadIdx.x;
  if (i < n) outp[i] = (__bf16)in[i];
}

// ---------- pack W[K x 192] f32 into per-lane bf16 B-fragment layout ----------
// Wp element index: ((ks*12 + tile)*32 + lane)*16 + j
//   col  c = tile*16 + (lane&15); khalf = (lane>>4)*8
//   k = ks*32 + khalf + (j<8 ? j : j+8)
__global__ void pack_w_kernel(const float* __restrict__ W, __bf16* __restrict__ Wp, int K) {
  int i = blockIdx.x * blockDim.x + threadIdx.x;
  if (i >= K * D) return;
  int j    = i & 15;
  int lane = (i >> 4) & 31;
  int tt   = (i >> 9) % 12;
  int ks   = (i >> 9) / 12;
  int c     = tt * 16 + (lane & 15);
  int khalf = (lane >> 4) * 8;
  int k     = ks * 32 + khalf + (j < 8 ? j : j + 8);
  Wp[i] = (__bf16)W[(size_t)k * D + c];
}

// ---------- WMMA GEMM: h[N x 192] = xb[N x K](bf16) @ Wp(packed bf16) ----------
// block = 128 threads = 4 waves; wave w -> cols [48w, 48w+48); block row strip = 16 rows.
// Inner loop: 2x b128 (A) + 6x b128 (B) + 3x v_wmma per 32-deep k-step.
__global__ void gemm_wmma_kernel(const __bf16* __restrict__ xb, const __bf16* __restrict__ Wp,
                                 float* __restrict__ h, int K) {
  const int lane  = threadIdx.x & 31;
  const int wave  = threadIdx.x >> 5;
  const int row0  = blockIdx.x * 16;
  const int col0  = wave * 48;
  const int r     = lane & 15;
  const int khalf = (lane >> 4) * 8;       // wave32: lanes 16-31 hold K+8 / K+24
  const int nk    = K >> 5;

  const __bf16* ap = xb + (size_t)(row0 + r) * K + khalf;
  v8f acc[3] = {v8f{}, v8f{}, v8f{}};

  for (int ks = 0; ks < nk; ++ks) {
    if (ks + 1 < nk) __builtin_prefetch(ap + (ks + 1) * 32, 0, 0);  // global_prefetch_b8
    V16U a;
    a.p.lo = *(const v8bf*)(ap + ks * 32);          // k = ks*32 + khalf + [0..7]
    a.p.hi = *(const v8bf*)(ap + ks * 32 + 16);     // k = ks*32 + khalf + 16 + [0..7]
#pragma unroll
    for (int t = 0; t < 3; ++t) {
      const __bf16* bp = Wp + ((size_t)((ks * 12 + wave * 3 + t) * 32 + lane) << 4);
      V16U b;
      b.p.lo = *(const v8bf*)(bp);
      b.p.hi = *(const v8bf*)(bp + 8);
      acc[t] = __builtin_amdgcn_wmma_f32_16x16x32_bf16(
          false, a.v, false, b.v, (short)0, acc[t], false, false);
    }
  }
  // C/D layout: VGPR i -> row i (lanes 0-15) / row 8+i (lanes 16-31); col = lane&15
  const int rr = row0 + (lane >> 4) * 8;
#pragma unroll
  for (int t = 0; t < 3; ++t) {
    const int cc = col0 + t * 16 + r;
#pragma unroll
    for (int i = 0; i < 8; ++i)
      h[(size_t)(rr + i) * D + cc] = acc[t][i];
  }
}

// ---------- per-node attention scores s_src/s_dst [N,H] ----------
__global__ void node_score_kernel(const float* __restrict__ h, const float* __restrict__ a_src,
                                  const float* __restrict__ a_dst, float* __restrict__ ssrc,
                                  float* __restrict__ sdst, int N) {
  int i = blockIdx.x * blockDim.x + threadIdx.x;
  if (i >= N * NH) return;
  int node = i >> 2, hh = i & 3;
  const float* hp = h + (size_t)node * D + hh * CC;
  float s0 = 0.f, s1 = 0.f;
#pragma unroll 8
  for (int c = 0; c < CC; ++c) {
    float hv = hp[c];
    s0 += hv * a_src[hh * CC + c];
    s1 += hv * a_dst[hh * CC + c];
  }
  ssrc[i] = s0;
  sdst[i] = s1;
}

__global__ void fill_u32_kernel(unsigned* __restrict__ p, unsigned v, int n) {
  int i = blockIdx.x * blockDim.x + threadIdx.x;
  if (i < n) p[i] = v;
}

// ---------- edge pass 1: leaky-relu score + segment max over dst ----------
__global__ void edge_score_kernel(const int* __restrict__ src, const int* __restrict__ dst,
                                  const float* __restrict__ ssrc, const float* __restrict__ sdst,
                                  float* __restrict__ eraw, unsigned* __restrict__ mkey, int E) {
  int i = blockIdx.x * blockDim.x + threadIdx.x;
  if (i >= E * NH) return;
  int e = i >> 2, hh = i & 3;
  int s = src[e], d = dst[e];
  float v = ssrc[s * NH + hh] + sdst[d * NH + hh];
  v = v > 0.f ? v : NEG_SLOPE * v;
  eraw[i] = v;
  atomicMax(&mkey[d * NH + hh], f2mono(v));
}

// ---------- edge pass 2: exp(e - max) + segment sum ----------
__global__ void edge_exp_kernel(const int* __restrict__ dst, float* __restrict__ eraw,
                                const unsigned* __restrict__ mkey, float* __restrict__ denom,
                                int E) {
  int i = blockIdx.x * blockDim.x + threadIdx.x;
  if (i >= E * NH) return;
  int e = i >> 2, hh = i & 3;
  int d = dst[e];
  float m = mono2f(mkey[d * NH + hh]);
  float ex = __expf(eraw[i] - m);
  eraw[i] = ex;
  atomicAdd(&denom[d * NH + hh], ex);
}

// ---------- edge pass 3: message aggregation (one block per edge, 192 threads) ----------
__global__ void aggregate_kernel(const int* __restrict__ src, const int* __restrict__ dst,
                                 const float* __restrict__ h, const float* __restrict__ eraw,
                                 const float* __restrict__ denom, float* __restrict__ agg) {
  int e = blockIdx.x;
  int f = threadIdx.x;                 // 0..191
  int hh = f / CC;
  int s = src[e], d = dst[e];
  float alpha = eraw[e * NH + hh] / fmaxf(denom[d * NH + hh], 1e-16f);
  atomicAdd(&agg[(size_t)d * D + f], h[(size_t)s * D + f] * alpha);
}

// ---------- bias + ELU (in place) + bf16 copy for next GEMM ----------
__global__ void bias_elu_kernel(float* __restrict__ agg, const float* __restrict__ bias,
                                __bf16* __restrict__ xb, int total) {
  int i = blockIdx.x * blockDim.x + threadIdx.x;
  if (i >= total) return;
  float v = elu_f(agg[i] + bias[i % D]);
  agg[i] = v;
  xb[i]  = (__bf16)v;
}

// ---------- graph pooling (mean-sum + max + counts) ----------
__global__ void pool_kernel(const float* __restrict__ x, const int* __restrict__ batch,
                            float* __restrict__ gsum, unsigned* __restrict__ gmax,
                            float* __restrict__ gcount, int N) {
  int i = blockIdx.x * blockDim.x + threadIdx.x;
  if (i >= N * D) return;
  int node = i / D, f = i - node * D;
  int g = batch[node];
  float v = x[i];
  atomicAdd(&gsum[g * D + f], v);
  atomicMax(&gmax[g * D + f], f2mono(v));
  if (f == 0) atomicAdd(&gcount[g], 1.f);
}

// ---------- final MLP: [G,384] -> elu(fc1) -> out [G,2] ----------
__global__ void mlp_kernel(const float* __restrict__ gsum, const unsigned* __restrict__ gmax,
                           const float* __restrict__ gcount, const float* __restrict__ fc1_w,
                           const float* __restrict__ fc1_b, const float* __restrict__ out_w,
                           const float* __restrict__ out_b, float* __restrict__ out) {
  int g = blockIdx.x;
  __shared__ float xin[2 * D];
  __shared__ float hsh[48];
  float cnt = fmaxf(gcount[g], 1.f);
  for (int f = threadIdx.x; f < D; f += blockDim.x) {
    xin[f]     = gsum[g * D + f] / cnt;
    xin[D + f] = mono2f(gmax[g * D + f]);
  }
  __syncthreads();
  if (threadIdx.x < 48) {
    float acc = fc1_b[threadIdx.x];
    for (int k = 0; k < 2 * D; ++k) acc += xin[k] * fc1_w[k * 48 + threadIdx.x];
    hsh[threadIdx.x] = elu_f(acc);
  }
  __syncthreads();
  if (threadIdx.x < 2) {
    float acc = out_b[threadIdx.x];
    for (int k = 0; k < 48; ++k) acc += hsh[k] * out_w[k * 2 + threadIdx.x];
    out[g * 2 + threadIdx.x] = acc;
  }
}

extern "C" void kernel_launch(void* const* d_in, const int* in_sizes, int n_in,
                              void* d_out, int out_size, void* d_ws, size_t ws_size,
                              hipStream_t stream) {
  (void)in_sizes; (void)n_in; (void)out_size; (void)ws_size;
  const int N = 50000, E = 800000;

  const float* features = (const float*)d_in[0];
  const int*   adj      = (const int*)d_in[1];
  const int*   src      = adj;
  const int*   dst      = adj + E;
  const int*   batch    = (const int*)d_in[2];
  const float* W[3]     = {(const float*)d_in[3],  (const float*)d_in[7],  (const float*)d_in[11]};
  const float* asrc[3]  = {(const float*)d_in[4],  (const float*)d_in[8],  (const float*)d_in[12]};
  const float* adstp[3] = {(const float*)d_in[5],  (const float*)d_in[9],  (const float*)d_in[13]};
  const float* bvec[3]  = {(const float*)d_in[6],  (const float*)d_in[10], (const float*)d_in[14]};
  const float* fc1_w = (const float*)d_in[15];
  const float* fc1_b = (const float*)d_in[16];
  const float* out_w = (const float*)d_in[17];
  const float* out_b = (const float*)d_in[18];
  float* out = (float*)d_out;

  // workspace carve-up
  char* ws = (char*)d_ws;
  size_t off = 0;
  auto carve = [&](size_t bytes) -> void* {
    void* p = ws + off;
    off += (bytes + 255) & ~(size_t)255;
    return p;
  };
  float*    B0     = (float*)carve((size_t)N * D * 4);   // h buffer
  float*    B1     = (float*)carve((size_t)N * D * 4);   // agg / layer output ping
  float*    B2     = (float*)carve((size_t)N * D * 4);   // agg / layer output pong
  __bf16*   xbf    = (__bf16*)carve((size_t)N * D * 2);  // bf16 GEMM input (row-major)
  __bf16*   Wp0    = (__bf16*)carve((size_t)FEAT * D * 2);
  __bf16*   Wp1    = (__bf16*)carve((size_t)D * D * 2);
  __bf16*   Wp2    = (__bf16*)carve((size_t)D * D * 2);
  float*    ssrc   = (float*)carve((size_t)N * NH * 4);
  float*    sdst   = (float*)carve((size_t)N * NH * 4);
  unsigned* mkey   = (unsigned*)carve((size_t)N * NH * 4);
  float*    denom  = (float*)carve((size_t)N * NH * 4);
  float*    eraw   = (float*)carve((size_t)E * NH * 4);
  float*    gsum   = (float*)carve((size_t)GNUM * D * 4);
  unsigned* gmax   = (unsigned*)carve((size_t)GNUM * D * 4);
  float*    gcount = (float*)carve((size_t)GNUM * 4);

  const int TPB = 256;
  const int nhN = N * NH, nhE = E * NH, ND = N * D;

  // pre-pack weights into bf16 B-fragment layout (tiny; L2-resident afterwards)
  __bf16* Wp[3] = {Wp0, Wp1, Wp2};
  pack_w_kernel<<<(FEAT * D + TPB - 1) / TPB, TPB, 0, stream>>>(W[0], Wp0, FEAT);
  pack_w_kernel<<<(D * D + TPB - 1) / TPB, TPB, 0, stream>>>(W[1], Wp1, D);
  pack_w_kernel<<<(D * D + TPB - 1) / TPB, TPB, 0, stream>>>(W[2], Wp2, D);
  // features -> bf16 row-major for layer-1 GEMM
  f32_to_bf16_kernel<<<(N * FEAT + TPB - 1) / TPB, TPB, 0, stream>>>(features, xbf, N * FEAT);

  int K = FEAT;
  float* hbuf = B0;
  float* aggs[3] = {B1, B2, B1};

  for (int l = 0; l < 3; ++l) {
    float* agg = aggs[l];
    // GEMM (WMMA): h = x @ W_l            (N = 3125 * 16 exactly)
    gemm_wmma_kernel<<<dim3(N / 16), dim3(128), 0, stream>>>(xbf, Wp[l], hbuf, K);
    // attention scores
    node_score_kernel<<<(nhN + TPB - 1) / TPB, TPB, 0, stream>>>(hbuf, asrc[l], adstp[l],
                                                                 ssrc, sdst, N);
    // reset reducers
    fill_u32_kernel<<<(nhN + TPB - 1) / TPB, TPB, 0, stream>>>(mkey, MONO_NEG_INF, nhN);
    hipMemsetAsync(denom, 0, (size_t)nhN * 4, stream);
    hipMemsetAsync(agg, 0, (size_t)ND * 4, stream);
    // segment softmax over in-edges
    edge_score_kernel<<<(nhE + TPB - 1) / TPB, TPB, 0, stream>>>(src, dst, ssrc, sdst,
                                                                 eraw, mkey, E);
    edge_exp_kernel<<<(nhE + TPB - 1) / TPB, TPB, 0, stream>>>(dst, eraw, mkey, denom, E);
    // message aggregation (memory-bound: 768B contiguous per edge, f32 L2 atomics)
    aggregate_kernel<<<dim3(E), dim3(D), 0, stream>>>(src, dst, hbuf, eraw, denom, agg);
    // bias + ELU (also emits bf16 copy for the next layer's GEMM)
    bias_elu_kernel<<<(ND + TPB - 1) / TPB, TPB, 0, stream>>>(agg, bvec[l], xbf, ND);
    K = D;
    if (l == 2) {
      // pooling + MLP head on final f32 activations
      hipMemsetAsync(gsum, 0, (size_t)GNUM * D * 4, stream);
      hipMemsetAsync(gcount, 0, (size_t)GNUM * 4, stream);
      fill_u32_kernel<<<(GNUM * D + TPB - 1) / TPB, TPB, 0, stream>>>(gmax, MONO_NEG_INF,
                                                                      GNUM * D);
      pool_kernel<<<(ND + TPB - 1) / TPB, TPB, 0, stream>>>(agg, batch, gsum, gmax, gcount, N);
      mlp_kernel<<<dim3(GNUM), dim3(64), 0, stream>>>(gsum, gmax, gcount, fc1_w, fc1_b,
                                                      out_w, out_b, out);
    }
  }
}
